// Attention_30966714204899
// MI455X (gfx1250) — compile-verified, matrix operand order checked by reading, not run on hardware
//
#include <hip/hip_runtime.h>

typedef __bf16 bf16_t;
typedef __attribute__((ext_vector_type(16))) __bf16 v16bf;
typedef __attribute__((ext_vector_type(8)))  __bf16 v8bf;
typedef __attribute__((ext_vector_type(8)))  float  v8f;

#define B_  2
#define S_  2048
#define D_  2048
#define H_  16
#define DH_ 128

// Load one 16-wide bf16 A/B fragment row-chunk: elements 0..7 at p, 8..15 at p+16.
// Caller passes p = rowBase + kBase + half*8 (per ISA 16-bit A layout).
__device__ __forceinline__ v16bf load_frag(const bf16_t* p) {
  v8bf lo = *(const v8bf*)(p);
  v8bf hi = *(const v8bf*)(p + 16);
  v16bf r;
#pragma unroll
  for (int i = 0; i < 8; ++i) { r[i] = lo[i]; r[i + 8] = hi[i]; }
  return r;
}

__device__ __forceinline__ v8f wmma_bf16(v16bf a, v16bf b, v8f c) {
  return __builtin_amdgcn_wmma_f32_16x16x32_bf16(false, a, false, b, (short)0, c,
                                                 false, false);
}

// ---------------- f32 -> bf16 convert ----------------
__global__ void __launch_bounds__(256) cvt_f32_bf16(const float* __restrict__ in,
                                                    bf16_t* __restrict__ out, int n) {
  int i = blockIdx.x * blockDim.x + threadIdx.x;
  int stride = gridDim.x * blockDim.x;
  for (; i < n; i += stride) out[i] = (bf16_t)in[i];
}

// ---------------- GEMM: C[M,N] = A[M,K] * B[N,K]^T (bf16 in, f32 out) ----------
// block: 256 thr = 8 waves arranged 2x4; wave -> 64 rows x 64 cols (16 WMMA tiles)
// Explicit double-buffer: loads for step k+32 issue before the WMMAs of step k,
// so every wait targets loads that are a full 16-WMMA step old.
__global__ void __launch_bounds__(256) gemm_bf16_nt(const bf16_t* __restrict__ A,
                                                    const bf16_t* __restrict__ Bw,
                                                    float* __restrict__ C,
                                                    int M, int N, int K) {
  const int w    = threadIdx.x >> 5;
  const int lane = threadIdx.x & 31;
  const int lr   = lane & 15;
  const int half = lane >> 4;
  const int wr = w >> 2;            // 0..1
  const int wc = w & 3;             // 0..3
  const int m0 = blockIdx.y * 128 + wr * 64;
  const int n0 = blockIdx.x * 256 + wc * 64;

  v8f acc[4][4];
#pragma unroll
  for (int mi = 0; mi < 4; ++mi)
#pragma unroll
    for (int ni = 0; ni < 4; ++ni) acc[mi][ni] = v8f{};

  const bf16_t* arow[4];
  const bf16_t* brow[4];
#pragma unroll
  for (int mi = 0; mi < 4; ++mi)
    arow[mi] = A + (size_t)(m0 + mi * 16 + lr) * K + half * 8;
#pragma unroll
  for (int ni = 0; ni < 4; ++ni)
    brow[ni] = Bw + (size_t)(n0 + ni * 16 + lr) * K + half * 8;

  // prologue: load first k-step fragments
  v16bf af0[4], bf0[4];
#pragma unroll
  for (int mi = 0; mi < 4; ++mi) af0[mi] = load_frag(arow[mi]);
#pragma unroll
  for (int ni = 0; ni < 4; ++ni) bf0[ni] = load_frag(brow[ni]);

  for (int k = 0; k < K - 32; k += 32) {
    // prefetch next k-step into the second buffer (issues as load clauses)
    v16bf af1[4], bf1[4];
#pragma unroll
    for (int mi = 0; mi < 4; ++mi) af1[mi] = load_frag(arow[mi] + k + 32);
#pragma unroll
    for (int ni = 0; ni < 4; ++ni) bf1[ni] = load_frag(brow[ni] + k + 32);

    // consume current buffer
#pragma unroll
    for (int ni = 0; ni < 4; ++ni)
#pragma unroll
      for (int mi = 0; mi < 4; ++mi)
        acc[mi][ni] = wmma_bf16(af0[mi], bf0[ni], acc[mi][ni]);

    // rotate
#pragma unroll
    for (int mi = 0; mi < 4; ++mi) af0[mi] = af1[mi];
#pragma unroll
    for (int ni = 0; ni < 4; ++ni) bf0[ni] = bf1[ni];
  }

  // epilogue k-step
#pragma unroll
  for (int ni = 0; ni < 4; ++ni)
#pragma unroll
    for (int mi = 0; mi < 4; ++mi)
      acc[mi][ni] = wmma_bf16(af0[mi], bf0[ni], acc[mi][ni]);

#pragma unroll
  for (int mi = 0; mi < 4; ++mi)
#pragma unroll
    for (int ni = 0; ni < 4; ++ni)
#pragma unroll
      for (int r = 0; r < 8; ++r)
        C[(size_t)(m0 + mi * 16 + r + 8 * half) * N + n0 + ni * 16 + lr] =
            acc[mi][ni][r];
}

// ---------------- RoPE + head split ----------------
// qkv f32 [B*S, 3*D] -> Qbf,Kbf [B*H][S][DH] bf16 (RoPE'd), Vt [B*H][DH][S] bf16
__global__ void __launch_bounds__(256) rope_split(const float* __restrict__ qkv,
                                                  bf16_t* __restrict__ qbf,
                                                  bf16_t* __restrict__ kbf,
                                                  bf16_t* __restrict__ vtbf) {
  int idx = blockIdx.x * blockDim.x + threadIdx.x;   // over B*S*D
  if (idx >= B_ * S_ * D_) return;
  int c = idx % D_;
  int tok = idx / D_;
  int s = tok % S_;
  int b = tok / S_;
  int h = c / DH_;
  int dim = c % DH_;

  const float* row = qkv + (size_t)tok * (3 * D_);
  float qv = row[c];
  float kv = row[D_ + c];
  float vv = row[2 * D_ + c];
  float qo = row[c ^ 64];            // rotate_half partner (dim +/- 64)
  float ko = row[D_ + (c ^ 64)];

  int j = dim & 63;
  float invf = __expf(-((float)(2 * j) / (float)DH_) * 9.210340371976184f); // ln(1e4)
  float ang = (float)s * invf;
  float cs = __cosf(ang), sn = __sinf(ang);
  float sgn = (dim < 64) ? -1.0f : 1.0f;
  float qr = qv * cs + sgn * qo * sn;
  float kr = kv * cs + sgn * ko * sn;

  size_t bh = (size_t)(b * H_ + h);
  qbf[(bh * S_ + s) * DH_ + dim] = (bf16_t)qr;
  kbf[(bh * S_ + s) * DH_ + dim] = (bf16_t)kr;
  vtbf[(bh * DH_ + dim) * S_ + s] = (bf16_t)vv;
}

// ---------------- Flash attention (causal) ----------------
// grid: (B*H, S/128); block 256 = 8 waves; wave -> 16 query rows, 32 keys/step
__global__ void __launch_bounds__(256) flash_attn(const bf16_t* __restrict__ Q,
                                                  const bf16_t* __restrict__ Kb,
                                                  const bf16_t* __restrict__ Vt,
                                                  bf16_t* __restrict__ Z) {
  __shared__ float Pt[8][16][32];   // per-wave private P transpose buffer (16 KB)

  const int w    = threadIdx.x >> 5;
  const int lane = threadIdx.x & 31;
  const int lr   = lane & 15;
  const int half = lane >> 4;
  const int bh = blockIdx.x;
  const int b = bh >> 4, h = bh & 15;
  const int q0 = blockIdx.y * 128 + w * 16;

  const bf16_t* Qh = Q  + (size_t)bh * S_ * DH_;
  const bf16_t* Kh = Kb + (size_t)bh * S_ * DH_;
  const bf16_t* Vh = Vt + (size_t)bh * DH_ * S_;

  // cache Q fragments for all 4 K-dim chunks (dh=128 -> 4 x k32)
  v16bf qf[4];
#pragma unroll
  for (int c = 0; c < 4; ++c)
    qf[c] = load_frag(Qh + (size_t)(q0 + lr) * DH_ + c * 32 + half * 8);

  v8f o[8];
#pragma unroll
  for (int t = 0; t < 8; ++t) o[t] = v8f{};
  float m_[8], l_[8];
#pragma unroll
  for (int r = 0; r < 8; ++r) { m_[r] = -1e30f; l_[r] = 0.0f; }

  const float scale = 0.08838834764831845f;  // 1/sqrt(128)
  const int qmax = q0 + 15;

  for (int kk = 0; kk <= qmax; kk += 32) {
    // S = Q * K^T  (two 16x16 score tiles = 32 keys); hoist frag loads per tile
    v8f s0 = v8f{}, s1 = v8f{};
    {
      v16bf kf[4];
#pragma unroll
      for (int c = 0; c < 4; ++c)
        kf[c] = load_frag(Kh + (size_t)(kk + lr) * DH_ + c * 32 + half * 8);
#pragma unroll
      for (int c = 0; c < 4; ++c) s0 = wmma_bf16(qf[c], kf[c], s0);
    }
    {
      v16bf kf[4];
#pragma unroll
      for (int c = 0; c < 4; ++c)
        kf[c] = load_frag(Kh + (size_t)(kk + 16 + lr) * DH_ + c * 32 + half * 8);
#pragma unroll
      for (int c = 0; c < 4; ++c) s1 = wmma_bf16(qf[c], kf[c], s1);
    }

    // online softmax, row-wise over 16 lanes (per C-layout: lane holds col N=lr)
    float p0[8], p1[8], fs[8];
#pragma unroll
    for (int r = 0; r < 8; ++r) {
      int row = q0 + r + 8 * half;
      float v0 = (kk + lr      <= row) ? s0[r] * scale : -1e30f;
      float v1 = (kk + 16 + lr <= row) ? s1[r] * scale : -1e30f;
      float tm = fmaxf(v0, v1);
      tm = fmaxf(tm, __shfl_xor(tm, 1, 32));
      tm = fmaxf(tm, __shfl_xor(tm, 2, 32));
      tm = fmaxf(tm, __shfl_xor(tm, 4, 32));
      tm = fmaxf(tm, __shfl_xor(tm, 8, 32));
      float mn = fmaxf(m_[r], tm);
      float f  = __expf(m_[r] - mn);
      float e0 = __expf(v0 - mn);
      float e1 = __expf(v1 - mn);
      float rs = e0 + e1;
      rs += __shfl_xor(rs, 1, 32);
      rs += __shfl_xor(rs, 2, 32);
      rs += __shfl_xor(rs, 4, 32);
      rs += __shfl_xor(rs, 8, 32);
      m_[r] = mn;
      l_[r] = l_[r] * f + rs;
      fs[r] = f; p0[r] = e0; p1[r] = e1;
    }
    // rescale running output
#pragma unroll
    for (int t = 0; t < 8; ++t)
#pragma unroll
      for (int r = 0; r < 8; ++r) o[t][r] *= fs[r];

    // transpose P (C-layout -> A-frag layout) through per-wave LDS tile
#pragma unroll
    for (int r = 0; r < 8; ++r) {
      Pt[w][r + 8 * half][lr]      = p0[r];
      Pt[w][r + 8 * half][lr + 16] = p1[r];
    }
    v16bf pa;
#pragma unroll
    for (int i = 0; i < 8; ++i) {
      pa[i]     = (bf16_t)Pt[w][lr][half * 8 + i];
      pa[i + 8] = (bf16_t)Pt[w][lr][16 + half * 8 + i];
    }

    // O += P * V (V transposed: row = dh index, contiguous keys); 2 groups of 4
#pragma unroll
    for (int g = 0; g < 2; ++g) {
      v16bf vb[4];
#pragma unroll
      for (int t = 0; t < 4; ++t)
        vb[t] = load_frag(Vh + (size_t)((g * 4 + t) * 16 + lr) * S_ + kk + half * 8);
#pragma unroll
      for (int t = 0; t < 4; ++t)
        o[g * 4 + t] = wmma_bf16(pa, vb[t], o[g * 4 + t]);
    }
  }

#pragma unroll
  for (int r = 0; r < 8; ++r) l_[r] = 1.0f / l_[r];
#pragma unroll
  for (int t = 0; t < 8; ++t)
#pragma unroll
    for (int r = 0; r < 8; ++r) {
      int row = q0 + r + 8 * half;
      Z[((size_t)(b * S_ + row)) * D_ + h * DH_ + t * 16 + lr] =
          (bf16_t)(o[t][r] * l_[r]);
    }
}

extern "C" void kernel_launch(void* const* d_in, const int* in_sizes, int n_in,
                              void* d_out, int out_size, void* d_ws, size_t ws_size,
                              hipStream_t stream) {
  const float* x    = (const float*)d_in[0];   // [2,2048,2048]
  const float* wqkv = (const float*)d_in[1];   // [6144,2048]
  const float* wo   = (const float*)d_in[2];   // [2048,2048]
  float* out = (float*)d_out;                  // [2,2048,2048]
  char* ws = (char*)d_ws;

  // workspace layout (bytes)
  bf16_t* xbf    = (bf16_t*)(ws + 0);           // 16 MiB
  bf16_t* wqkvbf = (bf16_t*)(ws + 16777216);    // 24 MiB
  bf16_t* wobf   = (bf16_t*)(ws + 41943040);    //  8 MiB
  float*  qkv    = (float*) (ws + 50331648);    // 96 MiB
  bf16_t* qbf    = (bf16_t*)(ws + 150994944);   // 16 MiB
  bf16_t* kbf    = (bf16_t*)(ws + 167772160);   // 16 MiB
  bf16_t* vtbf   = (bf16_t*)(ws + 184549376);   // 16 MiB
  bf16_t* zbf    = (bf16_t*)(ws + 201326592);   // 16 MiB
  if (ws_size < 218103808u) return;

  const int nX = B_ * S_ * D_;          // 8388608
  const int nWq = 3 * D_ * D_;          // 12582912
  const int nWo = D_ * D_;              // 4194304
  cvt_f32_bf16<<<(nX + 255) / 256, 256, 0, stream>>>(x, xbf, nX);
  cvt_f32_bf16<<<(nWq + 255) / 256, 256, 0, stream>>>(wqkv, wqkvbf, nWq);
  cvt_f32_bf16<<<(nWo + 255) / 256, 256, 0, stream>>>(wo, wobf, nWo);

  // QKV projection: [4096, 6144] = X[4096,2048] * Wqkv^T
  gemm_bf16_nt<<<dim3(6144 / 256, 4096 / 128), 256, 0, stream>>>(
      xbf, wqkvbf, qkv, B_ * S_, 3 * D_, D_);

  // RoPE + head split (+ V transpose)
  rope_split<<<(nX + 255) / 256, 256, 0, stream>>>(qkv, qbf, kbf, vtbf);

  // causal flash attention
  flash_attn<<<dim3(B_ * H_, S_ / 128), 256, 0, stream>>>(qbf, kbf, vtbf, zbf);

  // output projection: [4096, 2048] = Z[4096,2048] * Wo^T
  gemm_bf16_nt<<<dim3(2048 / 256, 4096 / 128), 256, 0, stream>>>(
      zbf, wobf, out, B_ * S_, D_, D_);
}
